// SimpleDynamicCurvatureMLP_50611894616641
// MI455X (gfx1250) — matrix operand order repaired and checked by reading, not compiled, not run on hardware
//
#include <hip/hip_runtime.h>
#include <math.h>

// ---------------- problem constants ----------------
#define BATCH   65536
#define IN_DIM  784
#define KPAD    800          // 784 padded to multiple of 32
#define HID     128
#define OUT_DIM 10
#define EPS     1e-7f
#define HSTRIDE 132          // f32 row stride for h/u tiles: 4*132 % 64 banks -> conflict-free
#define WSTRIDE 40           // bf16 row stride for weight slices: 20 dwords -> conflict-free b128

// ---------------- vector types ----------------
typedef __attribute__((ext_vector_type(16))) __bf16       v16bf;
typedef __attribute__((ext_vector_type(8)))  float        v8f;
typedef __attribute__((ext_vector_type(2)))  __bf16       bf16x2;
typedef __attribute__((ext_vector_type(4)))  float        f32x4;
typedef __attribute__((ext_vector_type(4)))  unsigned int u32x4;

union AFRAG { bf16x2 p[8]; v16bf bf; };
union BFRAG { u32x4  q[2]; v16bf bf; };

__device__ __forceinline__ unsigned short f2bf_u16(float f) {
    unsigned int b = __float_as_uint(f);
    unsigned int r = b + 0x7FFFu + ((b >> 16) & 1u);   // RNE
    return (unsigned short)(r >> 16);
}

// pack two floats into a bf16x2 using the hardware converter when available
__device__ __forceinline__ void pack2(AFRAG& a, int i, float lo, float hi) {
#if defined(__has_builtin) && __has_builtin(__builtin_amdgcn_cvt_pk_bf16_f32)
    auto t = __builtin_amdgcn_cvt_pk_bf16_f32(lo, hi);
    a.p[i] = __builtin_bit_cast(bf16x2, t);
#else
    a.p[i][0] = (__bf16)lo;
    a.p[i][1] = (__bf16)hi;
#endif
}

__device__ __forceinline__ float fast_tanh(float x) {
#if defined(__has_builtin) && __has_builtin(__builtin_amdgcn_tanhf)
    return __builtin_amdgcn_tanhf(x);
#else
    return tanhf(x);
#endif
}

__device__ __forceinline__ f32x4 ntload4(const float* p) {
    return __builtin_nontemporal_load((const f32x4*)p);
}

// ---------------- prep: transpose weights to bf16 [N][K] ----------------
__global__ void prep_w1(const float* __restrict__ w1, unsigned short* __restrict__ w1t) {
    int idx = blockIdx.x * 256 + threadIdx.x;          // HID * KPAD entries
    if (idx >= HID * KPAD) return;
    int n = idx / KPAD, k = idx - n * KPAD;
    float v = (k < IN_DIM) ? w1[(size_t)k * HID + n] : 0.0f;
    w1t[(size_t)n * KPAD + k] = f2bf_u16(v);
}

__global__ void prep_w2(const float* __restrict__ w2, unsigned short* __restrict__ w2t) {
    int idx = blockIdx.x * 256 + threadIdx.x;          // HID * HID entries
    if (idx >= HID * HID) return;
    int n = idx / HID, k = idx - n * HID;
    w2t[(size_t)n * HID + k] = f2bf_u16(w2[(size_t)k * HID + n]);
}

// cooperative copy of one K-slice (128 rows x 32 bf16) global -> LDS, 256 threads
__device__ __forceinline__ void stage_slice(const unsigned short* __restrict__ src,
                                            int rowStride, int k0,
                                            unsigned short* dst, int tid) {
    const int n    = tid >> 1;           // 0..127
    const int half = tid & 1;            // 16 bf16 (32B) per thread
    const u32x4* g = (const u32x4*)(src + (size_t)n * rowStride + k0 + half * 16);
    u32x4 q0 = g[0];
    u32x4 q1 = g[1];
    u32x4* d = (u32x4*)(dst + n * WSTRIDE + half * 16);
    d[0] = q0; d[1] = q1;
}

// B fragment (K=32 x N=16 tile column) from an LDS slice
__device__ __forceinline__ v16bf lds_bfrag(const unsigned short* slice, int n, int khalf) {
    BFRAG b;
    const unsigned short* p = slice + n * WSTRIDE + khalf * 8;
    b.q[0] = *(const u32x4*)p;
    b.q[1] = *(const u32x4*)(p + 16);
    return b.bf;
}

// ---------------- fused MLP + Poincare ball ----------------
__global__ __launch_bounds__(256) void fused_hyper_mlp(
    const float* __restrict__ x,       const float* __restrict__ b1,
    const float* __restrict__ b2,      const float* __restrict__ wout,
    const float* __restrict__ obias,   const float* __restrict__ curv,
    const unsigned short* __restrict__ w1t,
    const unsigned short* __restrict__ w2t,
    float* __restrict__ out)
{
    __shared__ float sh[128 * HSTRIDE];               // h tile, f32      (66 KB)
    __shared__ float su[128 * HSTRIDE];               // u tile -> z tile (66 KB)
    __shared__ unsigned short sw[2][128 * WSTRIDE];   // double-buffered weight slices (20 KB)

    const int tid     = threadIdx.x;
    const int wave    = tid >> 5;
    const int lane    = tid & 31;
    const int c16     = lane & 15;           // N column in tile / M row for A
    const int khalf   = lane >> 4;           // which K-half this lane owns
    const int rowBase = blockIdx.x * 128;    // batch rows handled by this block
    const int waveRow = wave * 16;           // M strip within block

    // ================= GEMM1: h = tanh(x @ W1 + b1) =================
    v8f acc[8] = {};
    const float* xrow = x + (size_t)(rowBase + waveRow + c16) * IN_DIM;

    stage_slice(w1t, KPAD, 0, sw[0], tid);
    __syncthreads();

    for (int kb = 0; kb < 25; ++kb) {
        const int cur = kb & 1;
        if (kb + 1 < 25) stage_slice(w1t, KPAD, (kb + 1) * 32, sw[cur ^ 1], tid);

        const int k0  = kb * 32;
        const int ka0 = k0 + khalf * 8;          // K chunk 0 (always < 784)
        const int ka1 = k0 + 16 + khalf * 8;     // K chunk 1 (pad region on kb==24)
        AFRAG a;
        {
            f32x4 q0 = ntload4(xrow + ka0);
            f32x4 q1 = ntload4(xrow + ka0 + 4);
            pack2(a, 0, q0[0], q0[1]); pack2(a, 1, q0[2], q0[3]);
            pack2(a, 2, q1[0], q1[1]); pack2(a, 3, q1[2], q1[3]);
        }
        if (kb < 24) {
            f32x4 q2 = ntload4(xrow + ka1);
            f32x4 q3 = ntload4(xrow + ka1 + 4);
            pack2(a, 4, q2[0], q2[1]); pack2(a, 5, q2[2], q2[3]);
            pack2(a, 6, q3[0], q3[1]); pack2(a, 7, q3[2], q3[3]);
        } else {
            pack2(a, 4, 0.f, 0.f); pack2(a, 5, 0.f, 0.f);
            pack2(a, 6, 0.f, 0.f); pack2(a, 7, 0.f, 0.f);
        }
        #pragma unroll
        for (int nt = 0; nt < 8; ++nt) {
            v16bf bf = lds_bfrag(sw[cur], nt * 16 + c16, khalf);
            acc[nt] = __builtin_amdgcn_wmma_f32_16x16x32_bf16(
                false, a.bf, false, bf, (short)0, acc[nt], false, false);
        }
        __syncthreads();
    }
    // bias + tanh -> LDS   (C/D layout: vgpr j, lanes<16 -> M=j, lanes>=16 -> M=j+8)
    #pragma unroll
    for (int nt = 0; nt < 8; ++nt) {
        const int col = nt * 16 + c16;
        const float bb = b1[col];
        #pragma unroll
        for (int j = 0; j < 8; ++j) {
            const int r = waveRow + khalf * 8 + j;
            sh[r * HSTRIDE + col] = fast_tanh(acc[nt][j] + bb);
        }
    }

    // ================= GEMM2: u = sigmoid(h @ W2 + b2) =================
    v8f acc2[8] = {};
    const float* hrow = &sh[(waveRow + c16) * HSTRIDE];

    stage_slice(w2t, HID, 0, sw[0], tid);
    __syncthreads();                    // covers sh writes + slice 0

    #pragma unroll
    for (int kb = 0; kb < 4; ++kb) {
        const int cur = kb & 1;
        if (kb + 1 < 4) stage_slice(w2t, HID, (kb + 1) * 32, sw[cur ^ 1], tid);

        const int ka0 = kb * 32 + khalf * 8;
        const int ka1 = kb * 32 + 16 + khalf * 8;
        AFRAG a;
        {
            f32x4 q0 = *(const f32x4*)(hrow + ka0);
            f32x4 q1 = *(const f32x4*)(hrow + ka0 + 4);
            f32x4 q2 = *(const f32x4*)(hrow + ka1);
            f32x4 q3 = *(const f32x4*)(hrow + ka1 + 4);
            pack2(a, 0, q0[0], q0[1]); pack2(a, 1, q0[2], q0[3]);
            pack2(a, 2, q1[0], q1[1]); pack2(a, 3, q1[2], q1[3]);
            pack2(a, 4, q2[0], q2[1]); pack2(a, 5, q2[2], q2[3]);
            pack2(a, 6, q3[0], q3[1]); pack2(a, 7, q3[2], q3[3]);
        }
        #pragma unroll
        for (int nt = 0; nt < 8; ++nt) {
            v16bf bf = lds_bfrag(sw[cur], nt * 16 + c16, khalf);
            acc2[nt] = __builtin_amdgcn_wmma_f32_16x16x32_bf16(
                false, a.bf, false, bf, (short)0, acc2[nt], false, false);
        }
        __syncthreads();
    }
    #pragma unroll
    for (int nt = 0; nt < 8; ++nt) {
        const int col = nt * 16 + c16;
        const float bb = b2[col];
        #pragma unroll
        for (int j = 0; j < 8; ++j) {
            const int r = waveRow + khalf * 8 + j;
            su[r * HSTRIDE + col] = 1.0f / (1.0f + expf(-(acc2[nt][j] + bb)));
        }
    }
    __syncthreads();

    // ================= Poincare ball layer (one thread per row) =================
    if (tid < 128) {
        float* hr = &sh[tid * HSTRIDE];
        float* ur = &su[tid * HSTRIDE];
        float hh = 0.f, uu = 0.f, hu = 0.f;
        for (int k = 0; k < HID; ++k) {
            const float hv = hr[k], uv = ur[k];
            hh = fmaf(hv, hv, hh); uu = fmaf(uv, uv, uu); hu = fmaf(hv, uv, hu);
        }
        const float s  = 1.0f / (1.0f + expf(-curv[0]));
        const float c  = 0.001f * (0.5f + 1.5f * s);
        const float sc = sqrtf(c);
        // mobius scalar factors: tanh(r*atanh(clamp(sc*||v||))) / (sc*||v||)
        const float nh   = sqrtf(fmaxf(hh, EPS * EPS));
        const float argh = fminf(fmaxf(sc * nh, EPS), 1.0f - 1e-5f);
        const float fh   = fast_tanh(0.3f * atanhf(argh)) / (sc * nh);
        const float nu   = sqrtf(fmaxf(uu, EPS * EPS));
        const float argu = fminf(fmaxf(sc * nu, EPS), 1.0f - 1e-5f);
        const float fu   = fast_tanh(0.7f * atanhf(argu)) / (sc * nu);
        // mobius add on scaled vectors (dot products scale by the scalar factors)
        const float uv2 = fh * fu * hu, uu2 = fh * fh * hh, vv2 = fu * fu * uu;
        const float An  = 1.0f + 2.0f * c * uv2 + c * vv2;
        const float Bn  = 1.0f - c * uu2;
        const float den = 1.0f + 2.0f * c * uv2 + c * c * uu2 * vv2;
        const float inv = 1.0f / fmaxf(den, EPS);
        float cA = An * fh * inv;          // z = cA*h + cB*u
        float cB = Bn * fu * inv;
        if (isnan(cA + cB)) { cA = 1.0f; cB = 0.0f; }   // NaN fallback: z = h
        for (int k = 0; k < HID; ++k) ur[k] = fmaf(cA, hr[k], cB * ur[k]);
    }
    __syncthreads();

    // ================= GEMM3: out = z @ Wout + obias  (N=10, VALU) =================
    {
        const int r  = tid >> 1;
        const int n0 = (tid & 1) * 5;
        const float* zr = &su[r * HSTRIDE];
        float o[5] = {0.f, 0.f, 0.f, 0.f, 0.f};
        for (int k = 0; k < HID; ++k) {
            const float zv = zr[k];
            #pragma unroll
            for (int n = 0; n < 5; ++n) o[n] = fmaf(zv, wout[k * OUT_DIM + n0 + n], o[n]);
        }
        float* op = out + (size_t)(rowBase + r) * OUT_DIM + n0;
        #pragma unroll
        for (int n = 0; n < 5; ++n) op[n] = o[n] + obias[n0 + n];
    }
}

// ---------------- launcher ----------------
extern "C" void kernel_launch(void* const* d_in, const int* in_sizes, int n_in,
                              void* d_out, int out_size, void* d_ws, size_t ws_size,
                              hipStream_t stream) {
    const float* x    = (const float*)d_in[0];
    const float* w1   = (const float*)d_in[1];
    const float* b1   = (const float*)d_in[2];
    const float* w2   = (const float*)d_in[3];
    const float* b2   = (const float*)d_in[4];
    const float* wout = (const float*)d_in[5];
    const float* ob   = (const float*)d_in[6];
    const float* curv = (const float*)d_in[7];
    float* out = (float*)d_out;

    unsigned short* w1t = (unsigned short*)d_ws;                       // 128*800*2 B
    unsigned short* w2t = (unsigned short*)((char*)d_ws + (size_t)HID * KPAD * 2);

    prep_w1<<<(HID * KPAD + 255) / 256, 256, 0, stream>>>(w1, w1t);
    prep_w2<<<(HID * HID + 255) / 256, 256, 0, stream>>>(w2, w2t);
    fused_hyper_mlp<<<BATCH / 128, 256, 0, stream>>>(x, b1, b2, wout, ob, curv, w1t, w2t, out);
}
